// SenseMemAct_46325517255182
// MI455X (gfx1250) — matrix operand order repaired and checked by reading, not compiled, not run on hardware
//
#include <hip/hip_runtime.h>
#include <hip/hip_bf16.h>

// ---------- types ----------
typedef __attribute__((ext_vector_type(16))) __bf16 v16bf;
typedef __attribute__((ext_vector_type(8)))  float  v8f;
typedef __attribute__((ext_vector_type(4)))  float  v4f;

union Frag {
    v16bf v;
    v4f   q[2];
};

// ---------- bf16 helpers (bit-level, no __bf16 arithmetic) ----------
__device__ __forceinline__ unsigned short f2bf(float f) {
    unsigned int u = __float_as_uint(f);
    unsigned int r = u + 0x7FFFu + ((u >> 16) & 1u);   // round-to-nearest-even
    return (unsigned short)(r >> 16);
}
__device__ __forceinline__ float bf2f(unsigned short h) {
    return __uint_as_float(((unsigned int)h) << 16);
}
__device__ __forceinline__ float sigf(float x) {
    return 1.0f / (1.0f + __expf(-x));
}

#define BDIM 64
#define LDIM 512
#define NDIM 128
#define MDIM 512
#define BT   16      // batch tile = WMMA M tile
#define NG   6       // gates: a,b,c,d,e,o

// ---------- f32 -> bf16 bulk convert ----------
__global__ void cvt_kernel(const float* __restrict__ src,
                           unsigned short* __restrict__ dst, int n) {
    int i = blockIdx.x * blockDim.x + threadIdx.x;
    if (i < n) dst[i] = f2bf(src[i]);
}

// ---------- composite weight: C[m][n] = sum_i Wi[m][i] * Ws[i][n] ----------
// Wi: [512,256] row-major, Ws: [256,128] row-major, C: [512,128] bf16
__global__ void composite_kernel(const float* __restrict__ Wi,
                                 const float* __restrict__ Ws,
                                 unsigned short* __restrict__ C) {
    int idx = blockIdx.x * blockDim.x + threadIdx.x;   // 512*128 threads
    int m = idx >> 7;
    int n = idx & 127;
    float s = 0.0f;
    #pragma unroll 8
    for (int i = 0; i < 256; ++i)
        s += Wi[m * 256 + i] * Ws[i * 128 + n];
    C[idx] = f2bf(s);
}

// ---------- the sequential scan: 4 blocks x 1024 threads (32 waves) ----------
// Block b owns batch rows [16b, 16b+16). Wave w owns M columns [16w, 16w+16).
// Per step: acc_g[16x16] = x_t @ C_g^T + hf @ Wh_g^T  (g=a..e; o input-only),
// then the in-register nonlinear state update. No inter-block communication.
__global__ __launch_bounds__(1024, 1) void scan_kernel(
    const unsigned short* __restrict__ xb,    // [64][512][128] bf16
    const unsigned short* __restrict__ Cw,    // [6][512][128]  bf16
    const unsigned short* __restrict__ Wh,    // [5][512][512]  bf16
    unsigned short* __restrict__ memb)        // [64][512][512] bf16 out
{
    __shared__ __align__(16) unsigned short hfb[BT * MDIM];   // 16 KB

    const int tid  = threadIdx.x;
    const int w    = tid >> 5;        // wave id 0..31 -> M col tile
    const int lane = tid & 31;
    const int kh   = lane >> 4;       // K-half select (A/B lane layout)
    const int ln   = lane & 15;
    const int Bbase = blockIdx.x * BT;

    // h0 = 0
    for (int i = tid; i < BT * MDIM; i += blockDim.x) hfb[i] = 0;
    __syncthreads();

    const int col  = w * 16 + ln;     // output column in M == weight row
    const int arow = ln;              // A-matrix row == local batch row

    const v8f vzero = {0.f, 0.f, 0.f, 0.f, 0.f, 0.f, 0.f, 0.f};
    v8f acc[NG];
    float hfv[8], hsv[8];             // persistent per-lane state (D layout)
    #pragma unroll
    for (int j = 0; j < 8; ++j) { hfv[j] = 0.f; hsv[j] = 0.f; }

    // Uniform zero offset, laundered through an SGPR each step. Addresses stay
    // rooted at the kernel-arg pointers (so they lower to global_load_b128,
    // not flat), but become loop-variant so the t-invariant weight fragments
    // cannot be hoisted out of the 512-step loop and spilled to scratch.
    size_t zoff = 0;

    const size_t cwBase = (size_t)col * NDIM + kh * 8;
    const size_t whBase = (size_t)col * MDIM + kh * 8;
    const size_t xBase  = (size_t)(Bbase + arow) * LDIM * NDIM + kh * 8;

    for (int t = 0; t < LDIM; ++t) {
        asm volatile("" : "+s"(zoff));

        #pragma unroll
        for (int g = 0; g < NG; ++g) acc[g] = vzero;

        // ---- input projection part: K = 128 over x_t with composites C ----
        const unsigned short* xrow = xb + xBase + (size_t)t * NDIM + zoff;
        #pragma unroll
        for (int kt = 0; kt < NDIM; kt += 32) {
            Frag A;
            A.q[0] = *(const v4f*)(xrow + kt);        // K k0..k0+7
            A.q[1] = *(const v4f*)(xrow + kt + 16);   // K k0+16..k0+23
            #pragma unroll
            for (int g = 0; g < NG; ++g) {
                Frag Bf;
                const unsigned short* brow =
                    Cw + cwBase + (size_t)g * MDIM * NDIM + kt + zoff;
                Bf.q[0] = *(const v4f*)(brow);
                Bf.q[1] = *(const v4f*)(brow + 16);
                acc[g] = __builtin_amdgcn_wmma_f32_16x16x32_bf16(
                    false, A.v, false, Bf.v, (short)0, acc[g], false, false);
            }
        }

        // ---- recurrent part: K = 512 over hf with Wh (gates a..e only) ----
        #pragma unroll 4
        for (int kt = 0; kt < MDIM; kt += 32) {
            Frag A;
            const unsigned short* hrow = hfb + arow * MDIM + kt + kh * 8;
            A.q[0] = *(const v4f*)(hrow);
            A.q[1] = *(const v4f*)(hrow + 16);
            #pragma unroll
            for (int g = 0; g < 5; ++g) {
                Frag Bf;
                const unsigned short* brow =
                    Wh + whBase + (size_t)g * MDIM * MDIM + kt + zoff;
                Bf.q[0] = *(const v4f*)(brow);
                Bf.q[1] = *(const v4f*)(brow + 16);
                acc[g] = __builtin_amdgcn_wmma_f32_16x16x32_bf16(
                    false, A.v, false, Bf.v, (short)0, acc[g], false, false);
            }
        }
        __syncthreads();   // all waves done reading hfb for this step

        // ---- in-register nonlinear state update (exclusive 16x16 region) ----
        #pragma unroll
        for (int j = 0; j < 8; ++j) {
            float pa = acc[0][j], pb = acc[1][j], pc = acc[2][j];
            float pd = acc[3][j], pe = acc[4][j], po = acc[5][j];
            float hf = hfv[j], hs = hsv[j];
            float a = 1.0f + tanhf(pa);
            float b = 1.5f * (1.0f + tanhf(pb));
            float c = 0.3f + 0.7f * sigf(pc);        // 3*DT + (1-3*DT)*sig
            float d = 0.03f * sigf(pd);              // 0.3*DT*sig
            float e = 1.0f + sigf(pe);
            float hfn = (1.0f - c) * hf +
                        c * tanhf(po + (a + b * hf * hf - hs) * hf);
            float t2 = e * hf; t2 *= t2; t2 *= t2;   // (e*hf)^4
            float hsn = hs * (1.0f - d) + d * t2;
            hfv[j] = hfn; hsv[j] = hsn;

            const int row = j + kh * 8;              // WMMA C/D lane layout
            unsigned short hb = f2bf(hfn);
            hfb[row * MDIM + col] = hb;
            memb[((size_t)(Bbase + row) * LDIM + t) * MDIM + col] = hb;
        }
        __syncthreads();   // hfb updated before next step's GEMM
    }
}

// ---------- act net + softmax over D=3 ----------
__global__ void act_kernel(const unsigned short* __restrict__ memb,
                           const float* __restrict__ Wa,
                           float* __restrict__ out, int nrows) {
    int r = blockIdx.x * blockDim.x + threadIdx.x;
    if (r >= nrows) return;
    const unsigned short* h = memb + (size_t)r * MDIM;
    float l0 = 0.f, l1 = 0.f, l2 = 0.f;
    #pragma unroll 4
    for (int m = 0; m < MDIM; ++m) {
        float hv = bf2f(h[m]);
        l0 += hv * Wa[0 * MDIM + m];
        l1 += hv * Wa[1 * MDIM + m];
        l2 += hv * Wa[2 * MDIM + m];
    }
    float mx = fmaxf(l0, fmaxf(l1, l2));
    float e0 = __expf(l0 - mx), e1 = __expf(l1 - mx), e2 = __expf(l2 - mx);
    float inv = 1.0f / (e0 + e1 + e2);
    out[(size_t)r * 3 + 0] = e0 * inv;
    out[(size_t)r * 3 + 1] = e1 * inv;
    out[(size_t)r * 3 + 2] = e2 * inv;
}

extern "C" void kernel_launch(void* const* d_in, const int* in_sizes, int n_in,
                              void* d_out, int out_size, void* d_ws, size_t ws_size,
                              hipStream_t stream) {
    // setup_inputs order:
    // 0 x, 1 Ws, 2 Wa, 3 Wia, 4 Wha, 5 Wib, 6 Whb, 7 Wic, 8 Whc,
    // 9 Wid, 10 Whd, 11 Wie, 12 Whe, 13 Wio
    const float* x  = (const float*)d_in[0];
    const float* Ws = (const float*)d_in[1];
    const float* Wa = (const float*)d_in[2];
    const float* Wi[6]  = { (const float*)d_in[3],  (const float*)d_in[5],
                            (const float*)d_in[7],  (const float*)d_in[9],
                            (const float*)d_in[11], (const float*)d_in[13] };
    const float* Whp[5] = { (const float*)d_in[4],  (const float*)d_in[6],
                            (const float*)d_in[8],  (const float*)d_in[10],
                            (const float*)d_in[12] };

    char* ws = (char*)d_ws;
    unsigned short* xb   = (unsigned short*)ws; ws += (size_t)BDIM * LDIM * NDIM * 2; // 8.4 MB
    unsigned short* Cw   = (unsigned short*)ws; ws += (size_t)6 * MDIM * NDIM * 2;    // 0.8 MB
    unsigned short* Whb  = (unsigned short*)ws; ws += (size_t)5 * MDIM * MDIM * 2;    // 2.6 MB
    unsigned short* memb = (unsigned short*)ws;                                        // 33.6 MB

    // bf16 conversions
    {
        int n = BDIM * LDIM * NDIM;
        cvt_kernel<<<(n + 255) / 256, 256, 0, stream>>>(x, xb, n);
    }
    for (int g = 0; g < 5; ++g) {
        int n = MDIM * MDIM;
        cvt_kernel<<<(n + 255) / 256, 256, 0, stream>>>(Whp[g], Whb + (size_t)g * n, n);
    }
    // composite input weights C_g = Wi_g @ Ws  (folds sense net away)
    for (int g = 0; g < 6; ++g) {
        int n = MDIM * NDIM;
        composite_kernel<<<(n + 255) / 256, 256, 0, stream>>>(Wi[g], Ws, Cw + (size_t)g * n);
    }
    // sequential WMMA scan: batch-parallel, sync-free across blocks
    scan_kernel<<<4, 1024, 0, stream>>>(xb, Cw, Whb, memb);
    // act + softmax
    {
        int n = BDIM * LDIM;
        act_kernel<<<(n + 255) / 256, 256, 0, stream>>>(memb, Wa, (float*)d_out, n);
    }
}
